// TalkingHeadAttention_87093346828552
// MI455X (gfx1250) — matrix operand order repaired and checked by reading, not compiled, not run on hardware
//
#include <hip/hip_runtime.h>
#include <hip/hip_bf16.h>

// ---------------------------------------------------------------------------
// Talking-heads attention for MI455X (gfx1250, wave32, WMMA).
//
// Pipeline (all bf16 WMMA f32-accum):
//   K_cvt   : f32 -> bf16 copies of x, Wqkv, Wout
//   K_gemm  : qkv[4096,3072] = xb @ Wqkvb^T + bqkv          (WMMA, ping-pong SW pipeline)
//   K_rope  : split qkv -> q,k (RoPE, bf16 [b,h,n,64]) and v^T (bf16 [b,h,64,n])
//   K_attn  : per (b, 16-query block), 16 waves = 16 heads:
//             pass0: stream j, QK^T (WMMA) -> LDS -> Wpre head-mix -> online m,Z
//             pass1: recompute, P=exp(..)*1/Z, Wpost head-mix, P@V (WMMA)
//   K_gemm  : out[4096,1024] = obuf @ Woutb^T + bout        (WMMA)
//
// Workspace: 100,663,296 bytes (96 MB), offsets below.
// ---------------------------------------------------------------------------

typedef __attribute__((ext_vector_type(16))) __bf16 v16bf;
typedef __attribute__((ext_vector_type(8)))  __bf16 bh8;
typedef __attribute__((ext_vector_type(8)))  float  v8f;

#define BATCH    2
#define SEQ      2048
#define DIM      1024
#define HEADS    16
#define HEAD_DIM 64
#define SCALE    0.125f   // 64^-0.5

static __device__ __forceinline__ v16bf join8(bh8 lo, bh8 hi) {
  v16bf r;
#pragma unroll
  for (int i = 0; i < 8; ++i) { r[i] = lo[i]; r[i + 8] = hi[i]; }
  return r;
}

// A fragment (16x32, bf16): lane holds row M = m0 + (lane&15).
// lanes 0-15 cover K = {k0+0..7, k0+16..23}; lanes 16-31 cover {k0+8..15, k0+24..31}.
static __device__ __forceinline__ v16bf load_a_frag(const __bf16* __restrict__ row,
                                                    int k0, int half) {
  bh8 lo = *(const bh8*)(row + k0 + 8 * half);
  bh8 hi = *(const bh8*)(row + k0 + 16 + 8 * half);
  return join8(lo, hi);
}

// B fragment (32x16, bf16): lane holds column N; same K order as A:
// K = {k0+8*half+0..7, k0+16+8*half+0..7} (column data = K-contiguous row of the [N,K] matrix).
static __device__ __forceinline__ v16bf load_b_frag(const __bf16* __restrict__ colrow,
                                                    int k0, int half) {
  bh8 lo = *(const bh8*)(colrow + k0 + 8 * half);
  bh8 hi = *(const bh8*)(colrow + k0 + 16 + 8 * half);
  return join8(lo, hi);
}

static __device__ __forceinline__ v8f wmma_bf16(v16bf a, v16bf b, v8f c) {
  return __builtin_amdgcn_wmma_f32_16x16x32_bf16(false, a, false, b,
                                                 (short)0, c, false, false);
}

// ---------------------------------------------------------------------------
__global__ __launch_bounds__(256) void cvt_f32_bf16(const float* __restrict__ in,
                                                    __bf16* __restrict__ out,
                                                    int n) {
  for (int i = blockIdx.x * blockDim.x + threadIdx.x; i < n;
       i += gridDim.x * blockDim.x)
    out[i] = (__bf16)in[i];
}

// ---------------------------------------------------------------------------
// C[M,N] = A[M,K](bf16) @ Bt[N,K]^T(bf16) + bias[N], f32 out.
// Block: 8 waves; wave w computes a 16(M) x 64(N) strip.
// Ping-pong double-buffered fragments: loads for chunk k+32 are in flight
// while WMMAs consume chunk k, so waits are partial (s_wait_loadcnt N, not 0).
// grid = (M/16, N/512); K must be a multiple of 64.
__global__ __launch_bounds__(256) void gemm_bias(const __bf16* __restrict__ A,
                                                 const __bf16* __restrict__ Bt,
                                                 const float* __restrict__ bias,
                                                 float* __restrict__ C,
                                                 int N, int K) {
  const int wave = threadIdx.x >> 5;
  const int lane = threadIdx.x & 31;
  const int half = lane >> 4;
  const int ln   = lane & 15;
  const int m0   = blockIdx.x * 16;
  const int n0   = (blockIdx.y * 8 + wave) * 64;

  const __bf16* arow = A + (size_t)(m0 + ln) * K;
  const __bf16* brow[4];
#pragma unroll
  for (int t = 0; t < 4; ++t) brow[t] = Bt + (size_t)(n0 + 16 * t + ln) * K;

  v8f acc[4];
#pragma unroll
  for (int t = 0; t < 4; ++t)
#pragma unroll
    for (int e = 0; e < 8; ++e) acc[t][e] = 0.f;

  // prologue: buffer 0 <- chunk 0
  v16bf a0 = load_a_frag(arow, 0, half);
  v16bf b0[4];
#pragma unroll
  for (int t = 0; t < 4; ++t) b0[t] = load_b_frag(brow[t], 0, half);

  for (int k0 = 0; k0 < K; k0 += 64) {
    // buffer 1 <- chunk k0+32 (in flight during buffer-0 WMMAs)
    v16bf a1 = load_a_frag(arow, k0 + 32, half);
    v16bf b1[4];
#pragma unroll
    for (int t = 0; t < 4; ++t) b1[t] = load_b_frag(brow[t], k0 + 32, half);
#pragma unroll
    for (int t = 0; t < 4; ++t) acc[t] = wmma_bf16(a0, b0[t], acc[t]);

    if (k0 + 64 < K) {       // buffer 0 <- chunk k0+64 (in flight during buffer-1 WMMAs)
      a0 = load_a_frag(arow, k0 + 64, half);
#pragma unroll
      for (int t = 0; t < 4; ++t) b0[t] = load_b_frag(brow[t], k0 + 64, half);
    }
#pragma unroll
    for (int t = 0; t < 4; ++t) acc[t] = wmma_bf16(a1, b1[t], acc[t]);
  }

  // C/D layout: element e -> row m0 + e + 8*half, col n0 + 16t + ln
#pragma unroll
  for (int t = 0; t < 4; ++t) {
    int n = n0 + 16 * t + ln;
    float bv = bias[n];
#pragma unroll
    for (int e = 0; e < 8; ++e)
      C[(size_t)(m0 + e + 8 * half) * N + n] = acc[t][e] + bv;
  }
}

// ---------------------------------------------------------------------------
// qkv f32 [b, n, 3072] -> qb,kb bf16 [b,h,n,64] with RoPE; vt bf16 [b,h,64,n].
// One thread per (b, pos, h, c) with c in 0..31 handling the (c, c+32) pair.
__global__ __launch_bounds__(256) void rope_pack(const float* __restrict__ qkv,
                                                 __bf16* __restrict__ qb,
                                                 __bf16* __restrict__ kb,
                                                 __bf16* __restrict__ vt) {
  int t = blockIdx.x * blockDim.x + threadIdx.x;   // 2^21 threads
  int c   = t & 31;
  int h   = (t >> 5) & 15;
  int pos = (t >> 9) & 2047;
  int b   = t >> 20;

  const float* src = qkv + ((size_t)(b * SEQ + pos)) * 3072 + h * 64 + c;
  // inv = theta^(-2c/64) = exp(-(c/32)*ln(1e4))
  float inv = __expf(-(float)c * (9.210340371976184f / 32.0f));
  float sn, cs;
  __sincosf((float)pos * inv, &sn, &cs);

  float q1 = src[0],    q2 = src[32];
  float k1 = src[1024], k2 = src[1024 + 32];
  float v1 = src[2048], v2 = src[2048 + 32];

  size_t qi = (((size_t)(b * HEADS + h)) * SEQ + pos) * HEAD_DIM + c;
  qb[qi]      = (__bf16)(q1 * cs - q2 * sn);
  qb[qi + 32] = (__bf16)(q1 * sn + q2 * cs);
  kb[qi]      = (__bf16)(k1 * cs - k2 * sn);
  kb[qi + 32] = (__bf16)(k1 * sn + k2 * cs);

  size_t vi = (((size_t)(b * HEADS + h)) * HEAD_DIM + c) * SEQ + pos;
  vt[vi]            = (__bf16)v1;
  vt[vi + 32 * SEQ] = (__bf16)v2;
}

// ---------------------------------------------------------------------------
// Fused attention. grid = BATCH * SEQ/16 blocks; 512 threads = 16 waves,
// wave w owns head w (both as producer of raw logits and as owner of mixed
// head w for softmax / output).
__global__ __launch_bounds__(512) void attn_kernel(const __bf16* __restrict__ qb,
                                                   const __bf16* __restrict__ kb,
                                                   const __bf16* __restrict__ vt,
                                                   const float* __restrict__ Wpre,
                                                   const float* __restrict__ Wpost,
                                                   __bf16* __restrict__ obuf) {
  __shared__ float  s_tiles[HEADS][32][8];   // per-head 16x16 f32 tile, (lane,e)
  __shared__ __bf16 s_p[HEADS][16][32];      // per-wave P'' staging, [i][j]

  const int w    = threadIdx.x >> 5;         // head / wave
  const int lane = threadIdx.x & 31;
  const int half = lane >> 4;
  const int ln   = lane & 15;
  const int b    = blockIdx.x >> 7;
  const int i0   = (blockIdx.x & 127) * 16;

  float wpre[HEADS], wpost[HEADS];
#pragma unroll
  for (int h = 0; h < HEADS; ++h) {
    wpre[h]  = Wpre[w * HEADS + h];
    wpost[h] = Wpost[w * HEADS + h];
  }

  const __bf16* qrow = qb + (((size_t)(b * HEADS + w)) * SEQ + i0 + ln) * HEAD_DIM;
  v16bf aq0 = load_a_frag(qrow, 0, half);
  v16bf aq1 = load_a_frag(qrow, 32, half);

  const __bf16* kb_h = kb + ((size_t)(b * HEADS + w)) * SEQ * HEAD_DIM;
  const __bf16* vt_h = vt + ((size_t)(b * HEADS + w)) * HEAD_DIM * SEQ;

  // Raw scaled logits for (all heads') j-chunk, then Wpre head-mix via LDS.
  auto compute_mixed = [&](int j0) -> v8f {
    const __bf16* krow = kb_h + (size_t)(j0 + ln) * HEAD_DIM;
    __builtin_prefetch(krow + 16 * HEAD_DIM, 0, 0);  // next chunk -> global_prefetch
    v16bf bk0 = load_b_frag(krow, 0, half);
    v16bf bk1 = load_b_frag(krow, 32, half);
    v8f L;
#pragma unroll
    for (int e = 0; e < 8; ++e) L[e] = 0.f;
    L = wmma_bf16(aq0, bk0, L);
    L = wmma_bf16(aq1, bk1, L);
    __syncthreads();                    // previous consumers of s_tiles done
#pragma unroll
    for (int e = 0; e < 8; ++e) s_tiles[w][lane][e] = L[e] * SCALE;
    __syncthreads();                    // all heads' raw tiles visible
    v8f mixed;
#pragma unroll
    for (int e = 0; e < 8; ++e) mixed[e] = 0.f;
#pragma unroll
    for (int h = 0; h < HEADS; ++h) {
      float g = wpre[h];
#pragma unroll
      for (int e = 0; e < 8; ++e) mixed[e] = fmaf(g, s_tiles[h][lane][e], mixed[e]);
    }
    return mixed;
  };

  // ---- pass 0: online (max, sum) per mixed-head row ----
  float m[8], z[8];
#pragma unroll
  for (int e = 0; e < 8; ++e) { m[e] = -INFINITY; z[e] = 0.f; }

  for (int j0 = 0; j0 < SEQ; j0 += 16) {
    v8f mixed = compute_mixed(j0);
#pragma unroll
    for (int e = 0; e < 8; ++e) {
      float v = mixed[e];
#pragma unroll
      for (int mask = 1; mask < 16; mask <<= 1) v = fmaxf(v, __shfl_xor(v, mask));
      float nm = fmaxf(m[e], v);
      float p = __expf(mixed[e] - nm);
#pragma unroll
      for (int mask = 1; mask < 16; mask <<= 1) p += __shfl_xor(p, mask);
      z[e] = z[e] * __expf(m[e] - nm) + p;
      m[e] = nm;
    }
  }
  float zinv[8];
#pragma unroll
  for (int e = 0; e < 8; ++e) zinv[e] = 1.0f / z[e];

  // ---- pass 1: P = exp(mixed-m)/Z, Wpost mix, P @ V ----
  v8f acc[4];
#pragma unroll
  for (int t = 0; t < 4; ++t)
#pragma unroll
    for (int e = 0; e < 8; ++e) acc[t][e] = 0.f;

  for (int j0 = 0; j0 < SEQ; j0 += 32) {
#pragma unroll
    for (int sub = 0; sub < 2; ++sub) {
      v8f mixed = compute_mixed(j0 + sub * 16);
      v8f P;
#pragma unroll
      for (int e = 0; e < 8; ++e) P[e] = __expf(mixed[e] - m[e]) * zinv[e];
      __syncthreads();                  // mixing reads of raw tiles done
#pragma unroll
      for (int e = 0; e < 8; ++e) s_tiles[w][lane][e] = P[e];
      __syncthreads();
      v8f a2;
#pragma unroll
      for (int e = 0; e < 8; ++e) a2[e] = 0.f;
#pragma unroll
      for (int g = 0; g < HEADS; ++g) {
        float wt = wpost[g];
#pragma unroll
        for (int e = 0; e < 8; ++e) a2[e] = fmaf(wt, s_tiles[g][lane][e], a2[e]);
      }
      // stage as bf16 [i][j] for the PV A-fragment
#pragma unroll
      for (int e = 0; e < 8; ++e)
        s_p[w][e + 8 * half][sub * 16 + ln] = (__bf16)a2[e];
    }
    __syncthreads();                    // own s_p writes complete (LDS in-order)

    // A = P''(16 x 32) from LDS; B = v^T rows (j-contiguous -> K-contiguous)
    bh8 plo = *(const bh8*)&s_p[w][ln][8 * half];
    bh8 phi = *(const bh8*)&s_p[w][ln][16 + 8 * half];
    v16bf ap = join8(plo, phi);
#pragma unroll
    for (int t = 0; t < 4; ++t) {
      const __bf16* vrow = vt_h + (size_t)(t * 16 + ln) * SEQ + j0;
      v16bf bv = load_b_frag(vrow, 0, half);
      acc[t] = wmma_bf16(ap, bv, acc[t]);
    }
  }

  // out[b, i, w*64 + d]  (bf16 for the final projection GEMM)
#pragma unroll
  for (int t = 0; t < 4; ++t)
#pragma unroll
    for (int e = 0; e < 8; ++e)
      obuf[(size_t)(b * SEQ + i0 + e + 8 * half) * DIM + w * 64 + t * 16 + ln] =
          (__bf16)acc[t][e];
}

// ---------------------------------------------------------------------------
extern "C" void kernel_launch(void* const* d_in, const int* in_sizes, int n_in,
                              void* d_out, int out_size, void* d_ws, size_t ws_size,
                              hipStream_t stream) {
  const float* x     = (const float*)d_in[0];
  const float* Wqkv  = (const float*)d_in[1];
  const float* bqkv  = (const float*)d_in[2];
  const float* Wpre  = (const float*)d_in[3];
  const float* Wpost = (const float*)d_in[4];
  const float* Wout  = (const float*)d_in[5];
  const float* bout  = (const float*)d_in[6];
  float* out = (float*)d_out;

  char* ws = (char*)d_ws;
  // 256-aligned workspace offsets; total = 100,663,296 bytes (96 MB)
  const size_t OFF_XB    = 0;          // 4096*1024 bf16      =  8 MB
  const size_t OFF_WQKVB = 8388608;    // 3072*1024 bf16      =  6 MB
  const size_t OFF_WOUTB = 14680064;   // 1024*1024 bf16      =  2 MB
  const size_t OFF_QKV   = 16777216;   // 4096*3072 f32       = 48 MB
  const size_t OFF_QB    = 67108864;   // 2*16*2048*64 bf16   =  8 MB
  const size_t OFF_KB    = 75497472;   //                        8 MB
  const size_t OFF_VT    = 83886080;   // transposed V           8 MB
  const size_t OFF_OB    = 92274688;   // 4096*1024 bf16         8 MB

  __bf16* xb     = (__bf16*)(ws + OFF_XB);
  __bf16* wqkvb  = (__bf16*)(ws + OFF_WQKVB);
  __bf16* woutb  = (__bf16*)(ws + OFF_WOUTB);
  float*  qkv    = (float*)(ws + OFF_QKV);
  __bf16* qb     = (__bf16*)(ws + OFF_QB);
  __bf16* kbuf   = (__bf16*)(ws + OFF_KB);
  __bf16* vt     = (__bf16*)(ws + OFF_VT);
  __bf16* obuf   = (__bf16*)(ws + OFF_OB);

  // 1) f32 -> bf16
  cvt_f32_bf16<<<4096, 256, 0, stream>>>(x,    xb,    BATCH * SEQ * DIM);
  cvt_f32_bf16<<<4096, 256, 0, stream>>>(Wqkv, wqkvb, 3 * DIM * DIM);
  cvt_f32_bf16<<<2048, 256, 0, stream>>>(Wout, woutb, DIM * DIM);

  // 2) qkv = xb @ Wqkv^T + bqkv   (M=4096, N=3072, K=1024)
  gemm_bias<<<dim3(256, 6), 256, 0, stream>>>(xb, wqkvb, bqkv, qkv, 3 * DIM, DIM);

  // 3) RoPE + head split + V transpose
  rope_pack<<<8192, 256, 0, stream>>>(qkv, qb, kbuf, vt);

  // 4) fused talking-heads attention
  attn_kernel<<<BATCH * (SEQ / 16), 512, 0, stream>>>(qb, kbuf, vt, Wpre, Wpost, obuf);

  // 5) out = obuf @ Wout^T + bout  (M=4096, N=1024, K=1024)
  gemm_bias<<<dim3(256, 2), 256, 0, stream>>>(obuf, woutb, bout, out, DIM, DIM);
}
